// PPRGo_43413529428215
// MI455X (gfx1250) — matrix operand order repaired and checked by reading, not compiled, not run on hardware
//
#include <hip/hip_runtime.h>
#include <hip/hip_bf16.h>

// ---------------- problem constants ----------------
#define NUM_EDGES   320000
#define FEAT        512
#define HID         512
#define NCLS        64
#define NNODES      10000

#define MT          64            // edge rows per block
#define ASTR        520           // LDS stride for activation tiles (bf16 elems), padded
#define WSTR        40            // LDS stride for weight K-chunk tiles (bf16 elems), padded
#define KCH         32            // K-chunk per WMMA sweep

typedef __attribute__((ext_vector_type(8)))  float    v8f;
typedef __attribute__((ext_vector_type(8)))  __bf16   v8bf;
typedef __attribute__((ext_vector_type(16))) __bf16   v16bf;
typedef __attribute__((ext_vector_type(4)))  uint32_t u32x4;
typedef __attribute__((ext_vector_type(8)))  uint32_t u32x8;

// A-fragment: 16-bit A 16x32 layout -> lane holds K = kb+{0..7} and kb+16+{0..7}
__device__ inline v16bf frag_split(const __bf16* p) {
    v8bf lo = *(const v8bf*)(p);
    v8bf hi = *(const v8bf*)(p + 16);
    return __builtin_shufflevector(lo, hi, 0,1,2,3,4,5,6,7,8,9,10,11,12,13,14,15);
}
// Unused: lane holds 16 consecutive K values
__device__ inline v16bf frag_contig(const __bf16* p) {
    v8bf lo = *(const v8bf*)(p);
    v8bf hi = *(const v8bf*)(p + 8);
    return __builtin_shufflevector(lo, hi, 0,1,2,3,4,5,6,7,8,9,10,11,12,13,14,15);
}

__device__ inline float relu1(float x) {            // single v_med3_f32
    return __builtin_amdgcn_fmed3f(x, 0.0f, __builtin_inff());
}

// ---- Tensor Data Mover: DMA a 2-D bf16 tile (rows x tile_w, row stride 512 elems)
// into LDS with hardware padding (pad_amount DWORD-units after 2^(pad_iv+1) DWORDs).
// D# per CDNA5 ISA; 2-group form (2-D tensor).
__device__ inline void tdm_load_tile(const __bf16* gsrc, const __bf16* ldst,
                                     uint32_t rows, uint32_t tile_w,
                                     uint32_t pad_iv, uint32_t pad_amt) {
    uint64_t ga = (uint64_t)(uintptr_t)gsrc;
    uint32_t la = (uint32_t)(uintptr_t)ldst;      // flat LDS addr truncates to offset
    u32x4 g0 = { 1u,                               // count=1 (valid user descriptor)
                 la,                               // lds_addr
                 (uint32_t)ga,                     // global_addr[31:0]
                 (uint32_t)((ga >> 32) & 0x01FFFFFFu) | 0x80000000u }; // [56:32] + type=2
    u32x8 g1 = { (1u << 16)                        // data_size = 2 bytes
               | (1u << 20)                        // pad_enable
               | (pad_iv << 22)                    // pad_interval
               | (pad_amt << 25),                  // pad_amount
                 512u << 16,                       // tensor_dim0 = 512 (lo16 @ bits 63:48)
                 rows << 16,                       // tensor_dim1 = rows (lo16 @ bits 95:80)
                 tile_w << 16,                     // tile_dim0 (bits 127:112)
                 rows,                             // tile_dim1 (bits 143:128)
                 512u,                             // tensor_dim0_stride (bits 191:160)
                 0u, 0u };                         // stride hi / tensor_dim1_stride = 0
    asm volatile("tensor_load_to_lds %0, %1" :: "s"(g0), "s"(g1) : "memory");
}

// ---------------- weight convert: fp32 W[k][n] -> bf16 W^T[n][k] ----------------
__global__ void pprgo_convert_w(const float* __restrict__ W0,
                                const float* __restrict__ W1,
                                const float* __restrict__ W2,
                                __bf16* __restrict__ wt0,
                                __bf16* __restrict__ wt1,
                                __bf16* __restrict__ wt2) {
    int idx = blockIdx.x * 256 + threadIdx.x;
    if (idx < 262144) {
        int n = idx >> 9, k = idx & 511;
        wt0[idx] = (__bf16)W0[k * 512 + n];
    } else if (idx < 524288) {
        int o = idx - 262144;
        int n = o >> 9, k = o & 511;
        wt1[o] = (__bf16)W1[k * 512 + n];
    } else if (idx < 557056) {
        int o = idx - 524288;
        int c = o >> 9, k = o & 511;
        wt2[o] = (__bf16)W2[k * 64 + c];
    }
}

__global__ void pprgo_zero_out(float* __restrict__ out) {
    int idx = blockIdx.x * 256 + threadIdx.x;
    if (idx < NNODES * NCLS) out[idx] = 0.0f;
}

// ---------------- fused 3-layer MLP + edge-weight + scatter-add ----------------
__global__ __launch_bounds__(256, 1)
void pprgo_fused(const float* __restrict__ x,
                 const int*   __restrict__ src,
                 const float* __restrict__ attr,
                 const __bf16* __restrict__ wt0,
                 const __bf16* __restrict__ wt1,
                 const __bf16* __restrict__ wt2,
                 float* __restrict__ out) {
    __shared__ __bf16 sA[MT * ASTR];         // x tile, later h1
    __shared__ __bf16 sH[MT * ASTR];         // h0
    __shared__ __bf16 sW[2][512 * WSTR];     // TDM double-buffered weight K-chunks
    __shared__ __bf16 sW2[NCLS * ASTR];      // resident W2^T [64][512] (padded)
    __shared__ float  sAttr[MT];
    __shared__ int    sSrc[MT];

    const int tid  = threadIdx.x;
    const int lane = tid & 31;
    const int w    = tid >> 5;       // wave 0..7 -> owns cols w*64..+63 (layers 0/1)
    const int eb   = blockIdx.x * MT;

    if (tid < MT) {
        sAttr[tid] = attr[eb + tid];
        sSrc[tid]  = src[eb + tid];
    }

    // TDM: start layer-0 chunk 0 + resident W2 while we stage the x tile
    if (w == 0) {
        tdm_load_tile(wt0, sW[0], 512, KCH, 3u, 3u);   // pad 16B per 64B row
        tdm_load_tile(wt2, sW2,  64, 512, 7u, 3u);     // pad 16B per 1024B row
    }

    // ---- load & cast x tile -> bf16 LDS ----
    #pragma unroll 4
    for (int v = 0; v < 32; ++v) {
        int g   = tid + v * 256;                 // each g covers 4 floats
        int row = g >> 7;
        int col = (g & 127) << 2;
        const float4 f = *(const float4*)(x + (size_t)(eb + row) * FEAT + col);
        __bf16* d = &sA[row * ASTR + col];
        d[0] = (__bf16)f.x; d[1] = (__bf16)f.y; d[2] = (__bf16)f.z; d[3] = (__bf16)f.w;
    }

    const int rlo   = lane & 15;
    const int khalf = lane >> 4;         // 0 or 1
    const int kbA   = khalf * 8;         // A-fragment K base within chunk

    const v8f vzero = {0.f,0.f,0.f,0.f,0.f,0.f,0.f,0.f};
    v8f acc[16];                         // acc[mt*4+nt] : 4 row-tiles x 4 col-tiles

    // ================= layers 0 and 1 (512 -> 512, relu) =================
    for (int layer = 0; layer < 2; ++layer) {
        const __bf16* Wt   = layer ? wt1 : wt0;
        const __bf16* Ain  = layer ? sH  : sA;
        __bf16*       Hout = layer ? sA  : sH;

        if (layer == 1 && w == 0) tdm_load_tile(wt1, sW[0], 512, KCH, 3u, 3u);

        #pragma unroll
        for (int i = 0; i < 16; ++i) acc[i] = vzero;

        for (int kc = 0; kc < 16; ++kc) {
            __syncthreads();                   // prev chunk consumed; buffers reusable
            if (w == 0) {
                if (kc < 15) {                 // DMA chunk kc+1 while computing kc
                    tdm_load_tile(Wt + (size_t)(kc + 1) * KCH, sW[(kc + 1) & 1],
                                  512, KCH, 3u, 3u);
                    // in-order completion; at layer0/kc0 the W2 DMA is also in flight
                    if (layer == 0 && kc == 0) __builtin_amdgcn_s_wait_tensorcnt(2);
                    else                       __builtin_amdgcn_s_wait_tensorcnt(1);
                } else {
                    __builtin_amdgcn_s_wait_tensorcnt(0);
                }
            }
            __syncthreads();                   // chunk kc visible
            const __bf16* sWb = sW[kc & 1];

            // 4 A-fragments (all 64 rows), 4 B-fragments (this wave's 64 cols)
            v16bf afr[4], bfr[4];
            #pragma unroll
            for (int mt = 0; mt < 4; ++mt)
                afr[mt] = frag_split(&Ain[(mt * 16 + rlo) * ASTR + kc * KCH + kbA]);
            #pragma unroll
            for (int nt = 0; nt < 4; ++nt)
                bfr[nt] = frag_split(&sWb[(w * 64 + nt * 16 + rlo) * WSTR + khalf * 8]);
            #pragma unroll
            for (int mt = 0; mt < 4; ++mt)
                #pragma unroll
                for (int nt = 0; nt < 4; ++nt)
                    acc[mt * 4 + nt] = __builtin_amdgcn_wmma_f32_16x16x32_bf16(
                        false, afr[mt], false, bfr[nt], (short)0, acc[mt * 4 + nt],
                        false, false);
        }
        __syncthreads();   // all sW / Ain reads done before overwriting Hout
        #pragma unroll
        for (int mt = 0; mt < 4; ++mt) {
            #pragma unroll
            for (int nt = 0; nt < 4; ++nt) {
                int n = w * 64 + nt * 16 + rlo;
                #pragma unroll
                for (int i = 0; i < 8; ++i) {
                    int r = mt * 16 + khalf * 8 + i;
                    Hout[r * ASTR + n] = (__bf16)relu1(acc[mt * 4 + nt][i]);
                }
            }
        }
    }
    __syncthreads();       // h1 (in sA) and resident W2 visible

    // ================= layer 2 (512 -> 64), barrier-free: W2 resident =================
    const int rg = w & 3;            // row group: rows rg*16..+15
    const int cg = w >> 2;           // col group: cols cg*32..+31
    v8f acc2[2];
    acc2[0] = vzero; acc2[1] = vzero;

    for (int kc = 0; kc < 16; ++kc) {
        v16bf afrag = frag_split(&sA[(rg * 16 + rlo) * ASTR + kc * KCH + kbA]);
        #pragma unroll
        for (int nt = 0; nt < 2; ++nt) {
            int n = cg * 32 + nt * 16 + rlo;
            v16bf bfrag = frag_split(&sW2[n * ASTR + kc * KCH + khalf * 8]);
            acc2[nt] = __builtin_amdgcn_wmma_f32_16x16x32_bf16(
                false, afrag, false, bfrag, (short)0, acc2[nt], false, false);
        }
    }

    // ---- epilogue: scale by edge_attr, scatter-add into out[src] ----
    #pragma unroll
    for (int nt = 0; nt < 2; ++nt) {
        int col = cg * 32 + nt * 16 + rlo;
        #pragma unroll
        for (int i = 0; i < 8; ++i) {
            int r = rg * 16 + khalf * 8 + i;
            float v = acc2[nt][i] * sAttr[r];
            atomicAdd(&out[(size_t)sSrc[r] * NCLS + col], v);
        }
    }
}

// ---------------- launcher ----------------
extern "C" void kernel_launch(void* const* d_in, const int* in_sizes, int n_in,
                              void* d_out, int out_size, void* d_ws, size_t ws_size,
                              hipStream_t stream) {
    const float* x    = (const float*)d_in[0];
    const int*   eidx = (const int*)d_in[1];     // [2][E]; row 0 = src
    const float* attr = (const float*)d_in[2];
    const float* W0   = (const float*)d_in[3];
    const float* W1   = (const float*)d_in[4];
    const float* W2   = (const float*)d_in[5];
    float* out = (float*)d_out;

    __bf16* wt0 = (__bf16*)d_ws;                 // [512][512] = W0^T
    __bf16* wt1 = wt0 + 512 * 512;               // [512][512] = W1^T
    __bf16* wt2 = wt1 + 512 * 512;               // [64][512]  = W2^T

    pprgo_convert_w<<<2176, 256, 0, stream>>>(W0, W1, W2, wt0, wt1, wt2);
    pprgo_zero_out<<<(NNODES * NCLS + 255) / 256, 256, 0, stream>>>(out);
    pprgo_fused<<<NUM_EDGES / MT, 256, 0, stream>>>(x, eidx, attr, wt0, wt1, wt2, out);
}
